// DocLSTM_7078106104356
// MI455X (gfx1250) — matrix operand order repaired and checked by reading, not compiled, not run on hardware
//
#include <hip/hip_runtime.h>
#include <hip/hip_bf16.h>

typedef __attribute__((ext_vector_type(16))) _Float16 v16h;
typedef __attribute__((ext_vector_type(8)))  _Float16 v8h;
typedef __attribute__((ext_vector_type(8)))  float    v8f;

#define NB 1025      // 1 head + 32*32 body sequences
#define NN 63        // nodes per tree
#define IND 300      // in_dim
#define MD 300       // M = 2*mem_dim
#define MEMD 150     // mem_dim
#define KPAD 320     // K=300 padded to multiple of 32
#define ROWS_X (NB*NN)        // 64575
#define ROWSPAD_X 64576       // multiple of 32

__device__ __forceinline__ float sigm(float x){ return 1.0f/(1.0f + expf(-x)); }
static inline unsigned cdiv(size_t a, size_t b){ return (unsigned)((a + b - 1)/b); }

// ---------------- WMMA fragment loaders (ISA 7.12.2 16-bit layouts) --------
// A 16x32 (MxK): lanes 0-15 M=lane, halves0-7 K=g*8+i, halves8-15 K=16+g*8+i
__device__ __forceinline__ v16h load_a_f16(const _Float16* __restrict__ X, int ld,
                                           int row, int kb, int g){
  const _Float16* p = X + (size_t)row*ld + kb + g*8;
  v8h lo = *(const v8h*)p;
  v8h hi = *(const v8h*)(p + 16);
  v16h a;
#pragma unroll
  for (int i = 0; i < 8; ++i){ a[i] = lo[i]; a[i+8] = hi[i]; }
  return a;
}
__device__ __forceinline__ v16h load_a_f32(const float* __restrict__ xr,
                                           int kb, int g, int K){
  v16h a;
#pragma unroll
  for (int i = 0; i < 8; ++i){
    int k0 = kb + g*8 + i, k1 = kb + 16 + g*8 + i;
    a[i]   = (k0 < K) ? (_Float16)xr[k0] : (_Float16)0.0f;
    a[i+8] = (k1 < K) ? (_Float16)xr[k1] : (_Float16)0.0f;
  }
  return a;
}
// B 32x16 (KxN): lane n=lane&15, g=lane>>4: halves0-7 K=g*8+i, halves8-15 K=16+g*8+i
__device__ __forceinline__ v16h load_b_f16(const _Float16* __restrict__ W, int ld,
                                           int col, int kb, int g){
  const _Float16* p = W + (size_t)col*ld + kb + g*8;
  v8h lo = *(const v8h*)p;
  v8h hi = *(const v8h*)(p + 16);
  v16h b;
#pragma unroll
  for (int i = 0; i < 8; ++i){ b[i] = lo[i]; b[i+8] = hi[i]; }
  return b;
}

// ---- register-blocked GEMM: wave computes 32x64 tile (MT=2 x NT=4) -------
// Y[r,n] = sum_k X[r,k]*W[n,k] + bias[n];  W is f16 padded [64*gridDim.y, KPAD]
template<bool AF32>
__global__ __launch_bounds__(32)
void gemm_b64(const void* __restrict__ Xv, int ldx,
              const _Float16* __restrict__ W, int ldw,
              const float* __restrict__ bias,
              float* __restrict__ Y, int ldy,
              int R, int K, int OUT, int ksteps){
  int lane = threadIdx.x, g = lane >> 4, m16 = lane & 15;
  int r0 = blockIdx.x * 32, n0 = blockIdx.y * 64;
  int row0 = r0 + m16, row1 = r0 + 16 + m16;
  const float*    Xf = (const float*)Xv;
  const _Float16* Xh = (const _Float16*)Xv;
  const float* xr0 = nullptr; const float* xr1 = nullptr;
  if (AF32){
    int c0 = row0 > R - 1 ? R - 1 : row0;
    int c1 = row1 > R - 1 ? R - 1 : row1;
    xr0 = Xf + (size_t)c0*ldx; xr1 = Xf + (size_t)c1*ldx;
  }
  v8f acc[2][4] = {};
  for (int ks = 0; ks < ksteps; ++ks){
    int kb = ks * 32;
    v16h a0, a1;
    if (AF32){ a0 = load_a_f32(xr0, kb, g, K); a1 = load_a_f32(xr1, kb, g, K); }
    else     { a0 = load_a_f16(Xh, ldx, row0, kb, g);
               a1 = load_a_f16(Xh, ldx, row1, kb, g); }
#pragma unroll
    for (int nt = 0; nt < 4; ++nt){
      v16h b = load_b_f16(W, ldw, n0 + nt*16 + m16, kb, g);
      acc[0][nt] = __builtin_amdgcn_wmma_f32_16x16x32_f16(false, a0, false, b,
                                                          (short)0, acc[0][nt], false, false);
      acc[1][nt] = __builtin_amdgcn_wmma_f32_16x16x32_f16(false, a1, false, b,
                                                          (short)0, acc[1][nt], false, false);
    }
  }
#pragma unroll
  for (int nt = 0; nt < 4; ++nt){
    int col = n0 + nt*16 + m16;
    if (col >= OUT) continue;
    float bv = bias[col];
#pragma unroll
    for (int mt = 0; mt < 2; ++mt){
#pragma unroll
      for (int r = 0; r < 8; ++r){
        int row = r0 + mt*16 + g*8 + r;
        if (row < R) Y[(size_t)row*ldy + col] = acc[mt][nt][r] + bv;
      }
    }
  }
}

// ---------------------------- prep kernels ---------------------------------
__global__ void zerof(float* p, size_t n){
  size_t i = (size_t)blockIdx.x*blockDim.x + threadIdx.x;
  if (i < n) p[i] = 0.0f;
}

__global__ void cvt_pad(const float* __restrict__ src, int rows, int cols,
                        _Float16* __restrict__ dst, int rowsPad, int colsPad){
  size_t idx = (size_t)blockIdx.x*blockDim.x + threadIdx.x;
  size_t tot = (size_t)rowsPad*colsPad;
  if (idx >= tot) return;
  int r = (int)(idx / colsPad), c = (int)(idx % colsPad);
  float v = (r < rows && c < cols) ? src[(size_t)r*cols + c] : 0.0f;
  dst[idx] = (_Float16)v;
}

__global__ void embed_f16(const float* __restrict__ emb, const int* __restrict__ head,
                          const int* __restrict__ body, _Float16* __restrict__ X){
  size_t idx = (size_t)blockIdx.x*blockDim.x + threadIdx.x;
  size_t tot = (size_t)ROWSPAD_X*KPAD;
  if (idx >= tot) return;
  int r = (int)(idx / KPAD), c = (int)(idx % KPAD);
  float v = 0.0f;
  if (r < ROWS_X && c < IND){
    int b = r / NN, n = r % NN;
    int tok = (b == 0) ? head[n] : body[(size_t)(b-1)*NN + n];
    v = emb[(size_t)tok*IND + c];
  }
  X[idx] = (_Float16)v;
}

__global__ void childsum(const float* __restrict__ cprev, float* __restrict__ csum,
                         int rows){
  size_t idx = (size_t)blockIdx.x*blockDim.x + threadIdx.x;
  size_t tot = (size_t)rows*MD;
  if (idx >= tot) return;
  int row = (int)(idx / MD), m = (int)(idx % MD);
  csum[idx] = cprev[((size_t)row*2 + 0)*MD + m] + cprev[((size_t)row*2 + 1)*MD + m];
}

// fused tree-LSTM node update (keeps upstream's child c/h swap quirk)
__global__ void tree_cell(const float* __restrict__ iou_x, const float* __restrict__ fxv,
                          const float* __restrict__ iou_h, const float* __restrict__ fh,
                          const float* __restrict__ hprev,
                          float* __restrict__ cout, float* __restrict__ hout,
                          int size, int off){
  size_t idx = (size_t)blockIdx.x*blockDim.x + threadIdx.x;
  size_t tot = (size_t)NB*size*MD;
  if (idx >= tot) return;
  int m = (int)(idx % MD);
  int row = (int)(idx / MD);            // b*size + j
  int b = row / size, j = row % size;
  size_t xr = (size_t)b*NN + off + j;
  float gi = iou_x[xr*900 + m]       + iou_h[(size_t)row*900 + m];
  float go = iou_x[xr*900 + 300 + m] + iou_h[(size_t)row*900 + 300 + m];
  float gu = iou_x[xr*900 + 600 + m] + iou_h[(size_t)row*900 + 600 + m];
  float i = sigm(gi), o = sigm(go), u = tanhf(gu);
  float fxm = fxv[xr*MD + m];
  float f0 = sigm(fh[((size_t)row*2 + 0)*MD + m] + fxm);
  float f1 = sigm(fh[((size_t)row*2 + 1)*MD + m] + fxm);
  float c = i*u + f0*hprev[((size_t)row*2 + 0)*MD + m]
                + f1*hprev[((size_t)row*2 + 1)*MD + m];
  cout[idx] = c;
  hout[idx] = o * tanhf(c);
}

// seq[s,p,:] = root_c[1 + p*32 + s], converted to f16, K padded
__global__ void gather_seq(const float* __restrict__ rootc, _Float16* __restrict__ X){
  size_t idx = (size_t)blockIdx.x*blockDim.x + threadIdx.x;
  if (idx >= (size_t)1024*KPAD) return;
  int r = (int)(idx / KPAD), c = (int)(idx % KPAD);
  float v = 0.0f;
  if (c < MD){ int s = r / 32, p = r % 32; v = rootc[(size_t)(1 + p*32 + s)*MD + c]; }
  X[idx] = (_Float16)v;
}

// ---------------- persistent per-paragraph LSTM (batch dim -> blocks) ------
__global__ __launch_bounds__(640)
void para_lstm(const float* __restrict__ xpart, const float* __restrict__ whh,
               const float* __restrict__ bhh, float* __restrict__ hT, int rev){
  __shared__ float h[MEMD], c[MEMD], gates[600];
  int p = blockIdx.x, t = threadIdx.x;
  if (t < MEMD){ h[t] = 0.0f; c[t] = 0.0f; }
  __syncthreads();
  for (int step = 0; step < 32; ++step){
    int s = rev ? (31 - step) : step;
    if (t < 600){
      float acc = xpart[(size_t)(s*32 + p)*600 + t] + bhh[t];
      const float* wr = whh + (size_t)t*MEMD;
      for (int k = 0; k < MEMD; ++k) acc += wr[k]*h[k];
      gates[t] = acc;
    }
    __syncthreads();
    if (t < MEMD){
      float i = sigm(gates[t]), f = sigm(gates[150 + t]);
      float gg = tanhf(gates[300 + t]), o = sigm(gates[450 + t]);
      float cn = f*c[t] + i*gg;
      c[t] = cn; h[t] = o*tanhf(cn);
    }
    __syncthreads();
  }
  if (t < MEMD) hT[(size_t)p*MEMD + t] = h[t];
}

// -------- tail: body scan (with upstream h-reuse quirk) + gated selection --
__global__ __launch_bounds__(640)
void tail(const float* __restrict__ hTf, const float* __restrict__ hTr,
          const float* __restrict__ wih, const float* __restrict__ whh,
          const float* __restrict__ bih, const float* __restrict__ bhh,
          const float* __restrict__ wsel, const float* __restrict__ bsel,
          const float* __restrict__ rootc, float* __restrict__ out){
  __shared__ float hsf[32*MEMD], hsb[32*MEMD];
  __shared__ float hf[MEMD], cf[MEMD], cb[MEMD], gates[600], logits[MD];
  __shared__ float smax, ssum;
  int t = threadIdx.x;
  if (t < MEMD){ hf[t] = 0.0f; cf[t] = 0.0f; cb[t] = 0.0f; }
  __syncthreads();
  for (int step = 0; step < 32; ++step){
    // forward cell: x_f = p_hidden[step] = [hTf[step], hTr[step]]
    if (t < 600){
      float acc = bih[t] + bhh[t];
      const float* wr = wih + (size_t)t*MD;
      const float* x0 = hTf + step*MEMD;
      const float* x1 = hTr + step*MEMD;
      for (int k = 0; k < MEMD; ++k) acc += wr[k]*x0[k];
      for (int k = 0; k < MEMD; ++k) acc += wr[150 + k]*x1[k];
      const float* wh = whh + (size_t)t*MEMD;
      for (int k = 0; k < MEMD; ++k) acc += wh[k]*hf[k];
      gates[t] = acc;
    }
    __syncthreads();
    if (t < MEMD){
      float i = sigm(gates[t]), f = sigm(gates[150 + t]);
      float gg = tanhf(gates[300 + t]), o = sigm(gates[450 + t]);
      float cn = f*cf[t] + i*gg; cf[t] = cn;
      float hn = o*tanhf(cn); hf[t] = hn; hsf[step*MEMD + t] = hn;
    }
    __syncthreads();
    // backward cell: x_b = p_hidden[31-step]; uses fresh hf (upstream quirk), c=cb
    if (t < 600){
      float acc = bih[t] + bhh[t];
      const float* wr = wih + (size_t)t*MD;
      int sb = 31 - step;
      const float* x0 = hTf + sb*MEMD;
      const float* x1 = hTr + sb*MEMD;
      for (int k = 0; k < MEMD; ++k) acc += wr[k]*x0[k];
      for (int k = 0; k < MEMD; ++k) acc += wr[150 + k]*x1[k];
      const float* wh = whh + (size_t)t*MEMD;
      for (int k = 0; k < MEMD; ++k) acc += wh[k]*hf[k];
      gates[t] = acc;
    }
    __syncthreads();
    if (t < MEMD){
      float i = sigm(gates[t]), f = sigm(gates[150 + t]);
      float gg = tanhf(gates[300 + t]), o = sigm(gates[450 + t]);
      float cn = f*cb[t] + i*gg; cb[t] = cn;
      hsb[step*MEMD + t] = o*tanhf(cn);
    }
    __syncthreads();
  }
  // selection: gate = softmax over feature axis (upstream Softmax(dim=1))
  float accOut = 0.0f;
  for (int p = 0; p < 32; ++p){
    if (t < MD){
      float acc = bsel[t];
      const float* wr = wsel + (size_t)t*600;
      const float* h0 = hsf + p*MEMD;   const float* h1 = hsb + p*MEMD;
      const float* r0 = hsf + 31*MEMD;  const float* r1 = hsb + 31*MEMD;
      for (int k = 0; k < MEMD; ++k) acc += wr[k]*h0[k];
      for (int k = 0; k < MEMD; ++k) acc += wr[150 + k]*h1[k];
      for (int k = 0; k < MEMD; ++k) acc += wr[300 + k]*r0[k];
      for (int k = 0; k < MEMD; ++k) acc += wr[450 + k]*r1[k];
      logits[t] = acc;
    }
    __syncthreads();
    if (t == 0){
      float mx = -1e30f;
      for (int j = 0; j < MD; ++j) mx = fmaxf(mx, logits[j]);
      float s = 0.0f;
      for (int j = 0; j < MD; ++j) s += expf(logits[j] - mx);
      smax = mx; ssum = s;
    }
    __syncthreads();
    if (t < MD){
      float gate = expf(logits[t] - smax)/ssum;
      float hb = (t < MEMD) ? hsf[p*MEMD + t] : hsb[p*MEMD + (t - MEMD)];
      accOut += hb*gate;
    }
    __syncthreads();
  }
  if (t < MD){ out[t] = accOut; out[MD + t] = rootc[t]; }  // rhidden = root_c[0]
}

// ------------------------------- launcher ----------------------------------
extern "C" void kernel_launch(void* const* d_in, const int* in_sizes, int n_in,
                              void* d_out, int out_size, void* d_ws, size_t ws_size,
                              hipStream_t stream){
  const float* emb     = (const float*)d_in[0];
  const int*   headTok = (const int*)  d_in[1];
  const int*   bodyTok = (const int*)  d_in[2];
  const float* w_ioux  = (const float*)d_in[3];
  const float* b_ioux  = (const float*)d_in[4];
  const float* w_iouh  = (const float*)d_in[5];
  const float* b_iouh  = (const float*)d_in[6];
  const float* w_fx    = (const float*)d_in[7];
  const float* b_fx    = (const float*)d_in[8];
  const float* w_fh    = (const float*)d_in[9];
  const float* b_fh    = (const float*)d_in[10];
  const float* w_ih_pf = (const float*)d_in[11];
  const float* w_hh_pf = (const float*)d_in[12];
  const float* b_ih_pf = (const float*)d_in[13];
  const float* b_hh_pf = (const float*)d_in[14];
  const float* w_ih_pr = (const float*)d_in[15];
  const float* w_hh_pr = (const float*)d_in[16];
  const float* b_ih_pr = (const float*)d_in[17];
  const float* b_hh_pr = (const float*)d_in[18];
  const float* w_ih_bd = (const float*)d_in[19];
  const float* w_hh_bd = (const float*)d_in[20];
  const float* b_ih_bd = (const float*)d_in[21];
  const float* b_hh_bd = (const float*)d_in[22];
  const float* w_sel   = (const float*)d_in[23];
  const float* b_sel   = (const float*)d_in[24];
  float* out = (float*)d_out;

  char* base = (char*)d_ws; size_t off = 0;
  auto take = [&](size_t bytes)->void*{
    void* p = base + off; off = (off + bytes + 255) & ~(size_t)255; return p;
  };
  _Float16* Xf16  = (_Float16*)take((size_t)ROWSPAD_X*KPAD*2);
  _Float16* Wioux = (_Float16*)take((size_t)960*KPAD*2);   // 900 -> 15 x 64
  _Float16* Wiouh = (_Float16*)take((size_t)960*KPAD*2);
  _Float16* Wfx   = (_Float16*)take((size_t)320*KPAD*2);   // 300 -> 5 x 64
  _Float16* Wfh   = (_Float16*)take((size_t)320*KPAD*2);
  _Float16* Wihpf = (_Float16*)take((size_t)640*KPAD*2);   // 600 -> 10 x 64
  _Float16* Wihpr = (_Float16*)take((size_t)640*KPAD*2);
  float* iou_x = (float*)take((size_t)ROWS_X*900*4);
  float* fx    = (float*)take((size_t)ROWS_X*300*4);
  float* cA    = (float*)take((size_t)NB*64*MD*4);
  float* hA    = (float*)take((size_t)NB*64*MD*4);
  float* cB    = (float*)take((size_t)NB*32*MD*4);
  float* hB    = (float*)take((size_t)NB*32*MD*4);
  float* csum  = (float*)take((size_t)NB*32*MD*4);
  float* iou_h = (float*)take((size_t)NB*32*900*4);
  float* fhbuf = (float*)take((size_t)NB*64*MD*4);
  _Float16* Xseq = (_Float16*)take((size_t)1024*KPAD*2);
  float* xpf   = (float*)take((size_t)1024*600*4);
  float* xpr   = (float*)take((size_t)1024*600*4);
  float* hTf   = (float*)take((size_t)32*MEMD*4);
  float* hTr   = (float*)take((size_t)32*MEMD*4);

  // leaf "children" h/c = zeros (fresh every call; graph-capture safe)
  size_t nz = (size_t)NB*64*MD;
  zerof<<<cdiv(nz,256),256,0,stream>>>(cA, nz);
  zerof<<<cdiv(nz,256),256,0,stream>>>(hA, nz);

  // weight convert + pad to f16 [OUTpad64, KPAD]
  cvt_pad<<<cdiv((size_t)960*KPAD,256),256,0,stream>>>(w_ioux,900,300,Wioux,960,KPAD);
  cvt_pad<<<cdiv((size_t)960*KPAD,256),256,0,stream>>>(w_iouh,900,300,Wiouh,960,KPAD);
  cvt_pad<<<cdiv((size_t)320*KPAD,256),256,0,stream>>>(w_fx,300,300,Wfx,320,KPAD);
  cvt_pad<<<cdiv((size_t)320*KPAD,256),256,0,stream>>>(w_fh,300,300,Wfh,320,KPAD);
  cvt_pad<<<cdiv((size_t)640*KPAD,256),256,0,stream>>>(w_ih_pf,600,300,Wihpf,640,KPAD);
  cvt_pad<<<cdiv((size_t)640*KPAD,256),256,0,stream>>>(w_ih_pr,600,300,Wihpr,640,KPAD);

  // embedding gather -> f16 [ROWSPAD_X, KPAD]
  embed_f16<<<cdiv((size_t)ROWSPAD_X*KPAD,256),256,0,stream>>>(emb, headTok, bodyTok, Xf16);

  // big WMMA GEMMs: iou_x and fx  (32x64 tile per wave)
  gemm_b64<false><<<dim3(ROWSPAD_X/32, 15),32,0,stream>>>(Xf16,KPAD,Wioux,KPAD,b_ioux,iou_x,900,ROWS_X,300,900,KPAD/32);
  gemm_b64<false><<<dim3(ROWSPAD_X/32,  5),32,0,stream>>>(Xf16,KPAD,Wfx,  KPAD,b_fx,  fx,   300,ROWS_X,300,300,KPAD/32);

  // tree levels 32,16,8,4,2,1
  float *cp = cA, *hp = hA, *co = cB, *ho = hB;
  int levoff = 0;
  for (int li = 0; li < 6; ++li){
    int size = 32 >> li;
    int rows = NB*size;
    childsum<<<cdiv((size_t)rows*MD,256),256,0,stream>>>(cp, csum, rows);
    gemm_b64<true><<<dim3(cdiv(rows,32),   15),32,0,stream>>>(csum,MD, Wiouh,KPAD, b_iouh, iou_h,900, rows,  300,900,KPAD/32);
    gemm_b64<true><<<dim3(cdiv(rows*2,32),  5),32,0,stream>>>(cp,  MD, Wfh,  KPAD, b_fh,   fhbuf,300, rows*2,300,300,KPAD/32);
    tree_cell<<<cdiv((size_t)rows*MD,256),256,0,stream>>>(iou_x, fx, iou_h, fhbuf, hp, co, ho, size, levoff);
    levoff += size;
    float* tc = cp; cp = co; co = tc;
    float* th = hp; hp = ho; ho = th;
  }
  // cp now holds root_c [1025, 300]; row 0 = rhidden

  // paragraph LSTMs
  gather_seq<<<cdiv((size_t)1024*KPAD,256),256,0,stream>>>(cp, Xseq);
  gemm_b64<false><<<dim3(1024/32, 10),32,0,stream>>>(Xseq,KPAD,Wihpf,KPAD,b_ih_pf,xpf,600,1024,300,600,KPAD/32);
  gemm_b64<false><<<dim3(1024/32, 10),32,0,stream>>>(Xseq,KPAD,Wihpr,KPAD,b_ih_pr,xpr,600,1024,300,600,KPAD/32);
  para_lstm<<<32,640,0,stream>>>(xpf, w_hh_pf, b_hh_pf, hTf, 0);
  para_lstm<<<32,640,0,stream>>>(xpr, w_hh_pr, b_hh_pr, hTr, 1);

  // body scan + gated selection + outputs
  tail<<<1,640,0,stream>>>(hTf, hTr, w_ih_bd, w_hh_bd, b_ih_bd, b_hh_bd,
                           w_sel, b_sel, cp, out);
}
